// BusemannPoincareMLR_4827543240762
// MI455X (gfx1250) — matrix operand order repaired and checked by reading, not compiled, not run on hardware
//
#include <hip/hip_runtime.h>

// Busemann-Poincare MLR, fused dual-GEMM + epilogue for MI455X (gfx1250).
// B=16384, K=2048, D=1024, C=1.0.
// Full path workspace: xbf(32MB) + pT,aT(8MB) + scalars(~80KB) ~= 40.1 MB.
// Fallback path workspace: ~8.1 MB (no xbf; x converted inline).

#define Bdim 16384
#define Kdim 2048
#define Ddim 1024

typedef __attribute__((ext_vector_type(16))) __bf16 v16bf;
typedef __attribute__((ext_vector_type(8)))  float  v8f;
typedef __attribute__((ext_vector_type(4)))  unsigned int v4u;
typedef __attribute__((ext_vector_type(4)))  float v4f;
typedef unsigned short u16;
typedef unsigned long long u64;

// f32 -> bf16 round-to-nearest-even
__device__ __forceinline__ u16 f2bf(float f) {
  unsigned u = __float_as_uint(f);
  u += 0x7FFFu + ((u >> 16) & 1u);
  return (u16)(u >> 16);
}

union Frag { v4u q[2]; v16bf v; };   // 32B: 16 bf16 = one WMMA A/B operand

__device__ __forceinline__ float block_reduce_256(float v, float* red) {
  const int t = threadIdx.x;
  red[t] = v; __syncthreads();
  for (int s = 128; s > 0; s >>= 1) {
    if (t < s) red[t] += red[t + s];
    __syncthreads();
  }
  float r = red[0];
  __syncthreads();
  return r;
}

// --- Stage 1: per-class prep: p = expmap0(point), a_hat = tangent/||tangent||,
//     pp[k]=||p||^2, pa[k]=<p,a_hat>; write bf16 transposed pT[d][k], aT[d][k].
__global__ __launch_bounds__(256)
void prep_classes(const float* __restrict__ point, const float* __restrict__ tangent,
                  u16* __restrict__ pT, u16* __restrict__ aT,
                  float* __restrict__ pp, float* __restrict__ pa)
{
  __shared__ float red[256];
  const int k = blockIdx.x;
  const int t = threadIdx.x;
  float pv[4], tv[4];
  float ssp = 0.f, sst = 0.f, dot = 0.f;
#pragma unroll
  for (int j = 0; j < 4; ++j) {
    const int d = t + j * 256;
    pv[j] = point[(size_t)k * Ddim + d];
    tv[j] = tangent[(size_t)k * Ddim + d];
    ssp += pv[j] * pv[j];
    sst += tv[j] * tv[j];
    dot += pv[j] * tv[j];
  }
  ssp = block_reduce_256(ssp, red);
  sst = block_reduce_256(sst, red);
  dot = block_reduce_256(dot, red);

  const float SQC = 1.0f;                     // sqrt(C), C = 1
  const float un = fmaxf(sqrtf(ssp), 1e-15f);
  const float scale = tanhf(SQC * un) / (SQC * un);
  const float an = fmaxf(sqrtf(sst), 1e-15f);
  const float ia = 1.0f / an;

  if (t == 0) {
    pp[k] = scale * scale * ssp;
    pa[k] = scale * ia * dot;
  }
#pragma unroll
  for (int j = 0; j < 4; ++j) {
    const int d = t + j * 256;
    pT[(size_t)d * Kdim + k] = f2bf(scale * pv[j]);
    aT[(size_t)d * Kdim + k] = f2bf(ia * tv[j]);
  }
}

// --- Stage 2: xx[b] = ||x_b||^2 ; optionally also emit bf16 copy of x.
template <bool WBF>
__global__ __launch_bounds__(256)
void row_norms(const float* __restrict__ x, float* __restrict__ xx,
               u16* __restrict__ xbf)
{
  __shared__ float red[256];
  const int b = blockIdx.x;
  const int t = threadIdx.x;
  float vv[4];
  float ss = 0.f;
#pragma unroll
  for (int j = 0; j < 4; ++j) {
    vv[j] = x[(size_t)b * Ddim + t + j * 256];
    ss += vv[j] * vv[j];
  }
  if (WBF) {
#pragma unroll
    for (int j = 0; j < 4; ++j)
      xbf[(size_t)b * Ddim + t + j * 256] = f2bf(vv[j]);
  }
  ss = block_reduce_256(ss, red);
  if (t == 0) xx[b] = ss;
}

// Shared epilogue: closed-form Mobius translation + Busemann distance (C=1).
__device__ __forceinline__ void epilogue_store(
    const v8f accP[2][2], const v8f accT[2][2],
    const float* __restrict__ pp, const float* __restrict__ pa,
    const float* __restrict__ xx, float* __restrict__ out,
    int bm, int bn, int wm, int wn, int r, int h)
{
  const float Cc = 1.0f, SQC = 1.0f;
#pragma unroll
  for (int i = 0; i < 2; ++i) {
    const int row0 = bm * 128 + wm * 32 + i * 16 + h * 8;
    float xarr[8];
    *(v4f*)&xarr[0] = *(const v4f*)(xx + row0);
    *(v4f*)&xarr[4] = *(const v4f*)(xx + row0 + 4);
#pragma unroll
    for (int j = 0; j < 2; ++j) {
      const int col = bn * 64 + wn * 32 + j * 16 + r;
      const float ppc = pp[col];
      const float pac = pa[col];
#pragma unroll
      for (int v = 0; v < 8; ++v) {
        const int row = row0 + v;
        const float xxv = xarr[v];
        const float xp = accP[i][j][v];   // <x, p>
        const float xa = accT[i][j][v];   // <x, a_hat>
        const float uv = -xp;
        const float alpha = 1.f + 2.f * Cc * uv + Cc * xxv;
        const float beta  = 1.f - Cc * ppc;
        const float gamma = 1.f + 2.f * Cc * uv + (Cc * Cc) * ppc * xxv;
        const float ig = 1.f / gamma;
        const float zz = (alpha * alpha * ppc + 2.f * alpha * beta * uv
                          + beta * beta * xxv) * ig * ig;
        const float za = (alpha * (-pac) + beta * xa) * ig;
        const float num = fmaxf(1.f - 2.f * SQC * za + Cc * zz, 1e-15f);
        const float den = fmaxf(1.f - Cc * zz, 1e-15f);
        out[(size_t)row * Kdim + col] = __logf(num / den) * (1.f / SQC);
      }
    }
  }
}

// --- Stage 3a (full path): all-bf16 operands, async-to-LDS staging,
//     k-step 64, double-buffered (2 x 32KB), one barrier per 16 WMMAs,
//     branch-free software pipeline (peeled prologue/epilogue).
// Block tile 128(M) x 64(N); 8 wave32s as 4(M) x 2(N).
__global__ __launch_bounds__(256)
void busemann_gemm_async(const u16* __restrict__ xbf,
                         const u16* __restrict__ pT, const u16* __restrict__ aT,
                         const float* __restrict__ pp, const float* __restrict__ pa,
                         const float* __restrict__ xx, float* __restrict__ out)
{
  // per buffer (u16 indices): X[0..8191] (16KB: 128 rows x 64),
  //                           P[8192..12287] (8KB: 64 rows x 64),
  //                           A[12288..16383] (8KB)
  __shared__ u16 lds[2][16384];   // 64 KB total

  const int t    = threadIdx.x;
  const int lane = t & 31;
  const int wave = t >> 5;
  const int wm   = wave & 3;
  const int wn   = wave >> 2;
  const int r    = lane & 15;
  const int h    = lane >> 4;
  const int bm   = blockIdx.x;   // 0..127
  const int bn   = blockIdx.y;   // 0..31

  const v8f vzero = {0.f,0.f,0.f,0.f,0.f,0.f,0.f,0.f};
  v8f accP[2][2], accT[2][2];
#pragma unroll
  for (int i = 0; i < 2; ++i)
#pragma unroll
    for (int j = 0; j < 2; ++j) { accP[i][j] = vzero; accT[i][j] = vzero; }

  // --- async staging: 16B chunks ---
  // X tile 16KB = 1024 chunks (8 per 128B row): thread owns c = t + 256*q, q=0..3
  //   row = c>>3, seg = c&7
  // P/A tiles 8KB = 512 chunks each: c = t + 256*q, q=0..1: row = c>>3, seg = c&7
  const unsigned ldsBase = (unsigned)(uintptr_t)&lds[0][0];
  const unsigned oX = (unsigned)(t * 16);            // + q*4096
  const unsigned oP = (unsigned)(16384 + t * 16);    // + q*4096
  const unsigned oA = (unsigned)(24576 + t * 16);    // + q*4096

  u64 gX = (u64)(uintptr_t)(xbf + (size_t)(bm * 128 + (t >> 3)) * Ddim + (t & 7) * 8);
  u64 gP = (u64)(uintptr_t)(pT + (size_t)(t >> 3) * Kdim + bn * 64 + (t & 7) * 8);
  u64 gA = (u64)(uintptr_t)(aT + (size_t)(t >> 3) * Kdim + bn * 64 + (t & 7) * 8);
  const u64 dXq = (u64)32 * Ddim * 2;   // +32 rows of x
  const u64 dPq = (u64)32 * Kdim * 2;   // +32 k-rows of pT/aT

  auto issue_async = [&](int buf) {
    const unsigned b = ldsBase + (unsigned)buf * 32768u;
#pragma unroll
    for (int q = 0; q < 4; ++q)
      asm volatile("global_load_async_to_lds_b128 %0, %1, off"
                   :: "v"(b + oX + (unsigned)q * 4096u), "v"(gX + (u64)q * dXq)
                   : "memory");
#pragma unroll
    for (int q = 0; q < 2; ++q)
      asm volatile("global_load_async_to_lds_b128 %0, %1, off"
                   :: "v"(b + oP + (unsigned)q * 4096u), "v"(gP + (u64)q * dPq)
                   : "memory");
#pragma unroll
    for (int q = 0; q < 2; ++q)
      asm volatile("global_load_async_to_lds_b128 %0, %1, off"
                   :: "v"(b + oA + (unsigned)q * 4096u), "v"(gA + (u64)q * dPq)
                   : "memory");
    gX += 128;                   // +64 bf16 along D
    gP += (u64)64 * Kdim * 2;    // +64 k-rows
    gA += (u64)64 * Kdim * 2;
  };

  auto compute = [&](int buf) {
    const u16* bX = &lds[buf][0];
    const u16* bP = &lds[buf][8192];
    const u16* bA = &lds[buf][12288];
#pragma unroll
    for (int ks = 0; ks < 2; ++ks) {
      Frag fX[2], fP[2], fT[2];
#pragma unroll
      for (int i = 0; i < 2; ++i) {
        const u16* base = bX + (wm * 32 + i * 16 + r) * 64 + ks * 32 + h * 8;
        fX[i].q[0] = *(const v4u*)(base);
        fX[i].q[1] = *(const v4u*)(base + 16);
      }
#pragma unroll
      for (int j = 0; j < 2; ++j) {
        const int nb = wn * 32 + j * 16;
        const u16* rowp = bP + (ks * 32 + h * 8) * 64 + nb + r;
        const u16* rowa = bA + (ks * 32 + h * 8) * 64 + nb + r;
#pragma unroll
        for (int e = 0; e < 4; ++e) {
          fP[j].q[0][e] = (unsigned)rowp[(2 * e) * 64] | ((unsigned)rowp[(2 * e + 1) * 64] << 16);
          fP[j].q[1][e] = (unsigned)rowp[(16 + 2 * e) * 64] | ((unsigned)rowp[(17 + 2 * e) * 64] << 16);
          fT[j].q[0][e] = (unsigned)rowa[(2 * e) * 64] | ((unsigned)rowa[(2 * e + 1) * 64] << 16);
          fT[j].q[1][e] = (unsigned)rowa[(16 + 2 * e) * 64] | ((unsigned)rowa[(17 + 2 * e) * 64] << 16);
        }
      }
#pragma unroll
      for (int i = 0; i < 2; ++i)
#pragma unroll
        for (int j = 0; j < 2; ++j) {
          accP[i][j] = __builtin_amdgcn_wmma_f32_16x16x32_bf16(
              false, fX[i].v, false, fP[j].v, (short)0, accP[i][j], false, false);
          accT[i][j] = __builtin_amdgcn_wmma_f32_16x16x32_bf16(
              false, fX[i].v, false, fT[j].v, (short)0, accT[i][j], false, false);
        }
    }
  };

#define WAIT_SYNC() do { \
    asm volatile("s_wait_asynccnt 0x0" ::: "memory"); \
    __syncthreads(); \
  } while (0)

  // software pipeline over 16 k-steps (D/64), static buffer indices
  issue_async(0);                 // step 0 -> buf0
  WAIT_SYNC();
  for (int s = 0; s < 14; s += 2) {
    issue_async(1);               // step s+1 -> buf1
    compute(0);                   // step s
    WAIT_SYNC();
    issue_async(0);               // step s+2 -> buf0
    compute(1);                   // step s+1
    WAIT_SYNC();
  }
  issue_async(1);                 // step 15 -> buf1
  compute(0);                     // step 14
  WAIT_SYNC();
  compute(1);                     // step 15
#undef WAIT_SYNC

  epilogue_store(accP, accT, pp, pa, xx, out, bm, bn, wm, wn, r, h);
}

// --- Stage 3b (fallback, small workspace): x converted f32->bf16 inline,
//     synchronous staging.
__global__ __launch_bounds__(256)
void busemann_gemm_conv(const float* __restrict__ x,
                        const u16* __restrict__ pT, const u16* __restrict__ aT,
                        const float* __restrict__ pp, const float* __restrict__ pa,
                        const float* __restrict__ xx, float* __restrict__ out)
{
  __shared__ u16 sX[128 * 32];
  __shared__ u16 sP[32 * 64];
  __shared__ u16 sA[32 * 64];

  const int t    = threadIdx.x;
  const int lane = t & 31;
  const int wave = t >> 5;
  const int wm   = wave & 3;
  const int wn   = wave >> 2;
  const int r    = lane & 15;
  const int h    = lane >> 4;
  const int bm   = blockIdx.x;
  const int bn   = blockIdx.y;

  const v8f vzero = {0.f,0.f,0.f,0.f,0.f,0.f,0.f,0.f};
  v8f accP[2][2], accT[2][2];
#pragma unroll
  for (int i = 0; i < 2; ++i)
#pragma unroll
    for (int j = 0; j < 2; ++j) { accP[i][j] = vzero; accT[i][j] = vzero; }

  const int xr4 = t >> 2;
  const int xc  = (t & 3) * 8;
  const int pr  = t >> 3;
  const int pc  = (t & 7) * 8;

  for (int kk = 0; kk < Ddim; kk += 32) {
#pragma unroll
    for (int pass = 0; pass < 2; ++pass) {
      const int row = pass * 64 + xr4;
      const float* src = x + (size_t)(bm * 128 + row) * Ddim + kk + xc;
      const v4f f0 = *(const v4f*)src;
      const v4f f1 = *(const v4f*)(src + 4);
      v4u o;
      o.x = (unsigned)f2bf(f0.x) | ((unsigned)f2bf(f0.y) << 16);
      o.y = (unsigned)f2bf(f0.z) | ((unsigned)f2bf(f0.w) << 16);
      o.z = (unsigned)f2bf(f1.x) | ((unsigned)f2bf(f1.y) << 16);
      o.w = (unsigned)f2bf(f1.z) | ((unsigned)f2bf(f1.w) << 16);
      *(v4u*)(sX + row * 32 + xc) = o;
    }
    {
      const size_t g = (size_t)(kk + pr) * Kdim + bn * 64 + pc;
      *(v4u*)(sP + pr * 64 + pc) = *(const v4u*)(pT + g);
      *(v4u*)(sA + pr * 64 + pc) = *(const v4u*)(aT + g);
    }
    __syncthreads();

    Frag fX[2], fP[2], fT[2];
#pragma unroll
    for (int i = 0; i < 2; ++i) {
      const u16* base = sX + (wm * 32 + i * 16 + r) * 32 + h * 8;
      fX[i].q[0] = *(const v4u*)(base);
      fX[i].q[1] = *(const v4u*)(base + 16);
    }
#pragma unroll
    for (int j = 0; j < 2; ++j) {
      const int nb = wn * 32 + j * 16;
      const u16* colp = sP + (h * 8) * 64 + nb + r;
      const u16* cola = sA + (h * 8) * 64 + nb + r;
#pragma unroll
      for (int e = 0; e < 4; ++e) {
        fP[j].q[0][e] = (unsigned)colp[(2 * e) * 64] | ((unsigned)colp[(2 * e + 1) * 64] << 16);
        fP[j].q[1][e] = (unsigned)colp[(16 + 2 * e) * 64] | ((unsigned)colp[(17 + 2 * e) * 64] << 16);
        fT[j].q[0][e] = (unsigned)cola[(2 * e) * 64] | ((unsigned)cola[(2 * e + 1) * 64] << 16);
        fT[j].q[1][e] = (unsigned)cola[(16 + 2 * e) * 64] | ((unsigned)cola[(17 + 2 * e) * 64] << 16);
      }
    }
#pragma unroll
    for (int i = 0; i < 2; ++i)
#pragma unroll
      for (int j = 0; j < 2; ++j) {
        accP[i][j] = __builtin_amdgcn_wmma_f32_16x16x32_bf16(
            false, fX[i].v, false, fP[j].v, (short)0, accP[i][j], false, false);
        accT[i][j] = __builtin_amdgcn_wmma_f32_16x16x32_bf16(
            false, fX[i].v, false, fT[j].v, (short)0, accT[i][j], false, false);
      }
    __syncthreads();
  }

  epilogue_store(accP, accT, pp, pa, xx, out, bm, bn, wm, wn, r, h);
}

extern "C" void kernel_launch(void* const* d_in, const int* in_sizes, int n_in,
                              void* d_out, int out_size, void* d_ws, size_t ws_size,
                              hipStream_t stream) {
  const float* x       = (const float*)d_in[0];   // [B,D]
  const float* point   = (const float*)d_in[1];   // [K,D]
  const float* tangent = (const float*)d_in[2];   // [K,D]
  float* out = (float*)d_out;                     // [B,K]

  const size_t szXbf = (size_t)Bdim * Ddim;       // u16 elements
  const size_t szPT  = (size_t)Ddim * Kdim;       // u16 elements
  const size_t need_full = (szXbf + 2 * szPT) * 2 + (2 * Kdim + Bdim) * 4 + 256;

  if (ws_size >= need_full) {
    u16*   xbf = (u16*)d_ws;                       // [B,D] bf16 row-major
    u16*   pT  = xbf + szXbf;                      // [D,K] bf16
    u16*   aT  = pT + szPT;                        // [D,K] bf16
    float* pp  = (float*)(aT + szPT);
    float* pa  = pp + Kdim;
    float* xx  = pa + Kdim;

    prep_classes<<<Kdim, 256, 0, stream>>>(point, tangent, pT, aT, pp, pa);
    row_norms<true><<<Bdim, 256, 0, stream>>>(x, xx, xbf);
    busemann_gemm_async<<<dim3(Bdim / 128, Kdim / 64), 256, 0, stream>>>(
        xbf, pT, aT, pp, pa, xx, out);
  } else {
    u16*   pT = (u16*)d_ws;
    u16*   aT = pT + szPT;
    float* pp = (float*)(aT + szPT);
    float* pa = pp + Kdim;
    float* xx = pa + Kdim;

    prep_classes<<<Kdim, 256, 0, stream>>>(point, tangent, pT, aT, pp, pa);
    row_norms<false><<<Bdim, 256, 0, stream>>>(x, xx, nullptr);
    busemann_gemm_conv<<<dim3(Bdim / 128, Kdim / 64), 256, 0, stream>>>(
        x, pT, aT, pp, pa, xx, out);
  }
}
